// Feature_loss_75179107549435
// MI455X (gfx1250) — matrix operand order repaired and checked by reading, not compiled, not run on hardware
//
#include <hip/hip_runtime.h>
#include <hip/hip_bf16.h>

typedef _Float16 v16h __attribute__((ext_vector_type(16)));
typedef _Float16 v8h  __attribute__((ext_vector_type(8)));
typedef float    v8f  __attribute__((ext_vector_type(8)));
typedef unsigned int v4u __attribute__((ext_vector_type(4)));
typedef int v8i __attribute__((ext_vector_type(8)));
typedef int v4i __attribute__((ext_vector_type(4)));

#define B_ROWS 4096
#define D_DIM  1024
#define C_DIM  1000
#define TEMP_INV 2.0f   // 1/0.5
#define LAMBD 0.05f
#define EPS 1e-8f

// ---------------------------------------------------------------- zero scratch
__global__ void k_zero(float* p, int n) {
    int i = blockIdx.x * blockDim.x + threadIdx.x;
    if (i < n) p[i] = 0.0f;
}

// ------------------------------------------------- row norms + f16 conversion
__global__ void k_normalize(const float* __restrict__ x, _Float16* __restrict__ xn,
                            int D) {
    __shared__ float red[256];
    __shared__ float s_rinv;
    int row = blockIdx.x;
    int tid = threadIdx.x;
    const float* xr = x + (size_t)row * D;
    float ss = 0.0f;
    for (int k = tid; k < D; k += 256) { float v = xr[k]; ss += v * v; }
    red[tid] = ss;
    __syncthreads();
    for (int s = 128; s > 0; s >>= 1) {
        if (tid < s) red[tid] += red[tid + s];
        __syncthreads();
    }
    if (tid == 0) {
        float nrm = sqrtf(red[0]);
        s_rinv = 1.0f / fmaxf(nrm, EPS);
    }
    __syncthreads();
    float rinv = s_rinv;
    _Float16* xo = xn + (size_t)row * D;
    for (int k = tid; k < D; k += 256) xo[k] = (_Float16)(xr[k] * rinv);
}

// -------------------------------------- first-match positive similarity S[i,j*]
__global__ void k_pos(const _Float16* __restrict__ xn, const int* __restrict__ y,
                      float* __restrict__ posS, int B, int D) {
    int tid  = threadIdx.x;
    int lane = tid & 31;
    int wave = tid >> 5;
    int i = blockIdx.x * 8 + wave;
    if (i >= B) return;
    int yi = y[i];
    int jstar = i;
    for (int jb = 0; jb < B; jb += 32) {
        bool pred = (y[jb + lane] == yi);
#if __has_builtin(__builtin_amdgcn_ballot_w32)
        unsigned mask = __builtin_amdgcn_ballot_w32(pred);
#else
        unsigned mask = (unsigned)__ballot(pred);
#endif
        if (mask != 0u) { jstar = jb + __builtin_ctz(mask); break; }
    }
    const _Float16* xi = xn + (size_t)i * D;
    const _Float16* xj = xn + (size_t)jstar * D;
    float s = 0.0f;
    for (int k = lane; k < D; k += 32) s += (float)xi[k] * (float)xj[k];
    for (int off = 16; off > 0; off >>= 1) s += __shfl_xor(s, off, 32);
    if (lane == 0) posS[i] = s;
}

// ---------------------------------------------- cross-entropy (atomic sum of NLL)
__global__ void k_ce(const float* __restrict__ data, const int* __restrict__ y,
                     float* __restrict__ ce_acc, int C) {
    __shared__ float red[256];
    int row = blockIdx.x;
    int tid = threadIdx.x;
    const float* dr = data + (size_t)row * C;
    float m = -3.4e38f;
    for (int c = tid; c < C; c += 256) m = fmaxf(m, dr[c]);
    red[tid] = m;
    __syncthreads();
    for (int s = 128; s > 0; s >>= 1) {
        if (tid < s) red[tid] = fmaxf(red[tid], red[tid + s]);
        __syncthreads();
    }
    float mx = red[0];
    __syncthreads();
    float sum = 0.0f;
    for (int c = tid; c < C; c += 256) sum += __expf(dr[c] - mx);
    red[tid] = sum;
    __syncthreads();
    for (int s = 128; s > 0; s >>= 1) {
        if (tid < s) red[tid] += red[tid + s];
        __syncthreads();
    }
    if (tid == 0) {
        float lse = mx + __logf(red[0]);
        float logp = dr[y[row]] - lse;
        atomicAdd(ce_acc, -logp);
    }
}

// ---------------------------------------------------------------- TDM helper
// DMA a [tile_d1 x tile_d0] f16 tile (row stride = stride_elems) from global
// memory into LDS at byte offset lds_off, padding each 64B row with 16B so the
// LDS row stride becomes 40 halves (matches LDS_STRIDE below).
__device__ __forceinline__ void tdm_load_tile_f16(unsigned lds_off,
                                                  const _Float16* gptr,
                                                  unsigned tensor_d0, unsigned tensor_d1,
                                                  unsigned tile_d0, unsigned tile_d1,
                                                  unsigned stride_elems) {
    unsigned long long ga = (unsigned long long)(uintptr_t)gptr;
    v4u g0;
    g0[0] = 1u;                                        // count=1 (valid, user mode)
    g0[1] = lds_off;                                   // LDS byte address
    g0[2] = (unsigned)(ga & 0xFFFFFFFFu);              // global_addr[31:0]
    g0[3] = (unsigned)((ga >> 32) & 0x01FFFFFFu) | (2u << 30); // addr[56:32] | type=2
    v8i g1;
    // data_size=2B (code 1), pad_enable, pad_interval=16 DWORDs (code 3),
    // pad_amount=4 DWORDs (code 3)
    g1[0] = (int)((1u << 16) | (1u << 20) | (3u << 22) | (3u << 25));
    g1[1] = (int)(tensor_d0 << 16);                            // dim0[15:0] @ [63:48]
    g1[2] = (int)((tensor_d0 >> 16) | (tensor_d1 << 16));      // dim0 hi | dim1 lo
    g1[3] = (int)((tensor_d1 >> 16) | (tile_d0 << 16));        // dim1 hi | tile_dim0
    g1[4] = (int)tile_d1;                                      // tile_dim1 (tile_dim2=0)
    g1[5] = (int)stride_elems;                                 // dim0_stride low 32
    g1[6] = 0;                                                 // stride hi | dim1_stride lo
    g1[7] = 0;
    v4i z = {0, 0, 0, 0};
#if defined(__clang_major__) && (__clang_major__ >= 23)
    v8i z8 = {0, 0, 0, 0, 0, 0, 0, 0};
    __builtin_amdgcn_tensor_load_to_lds(g0, g1, z, z, z8, 0);
#else
    __builtin_amdgcn_tensor_load_to_lds(g0, g1, z, z, 0);
#endif
}

// --------------------------------------------------------------- main GEMM
// Grid 32x32; block computes a 128x128 tile of S = xn*xn^T via WMMA f16 with
// TDM double-buffered LDS staging; fused epilogue: total[i] += sum_j exp(2*S).
#define LDS_STRIDE 40   // 32 + 8 halves TDM padding
#define TILE_H     (128 * LDS_STRIDE)
__global__ void
__launch_bounds__(256)
k_gemm_rowsum(const _Float16* __restrict__ xn, float* __restrict__ total,
              int B, int D) {
    __shared__ _Float16 As[2 * TILE_H];
    __shared__ _Float16 Bs[2 * TILE_H];

    int tid  = threadIdx.x;
    int lane = tid & 31;
    int wave = tid >> 5;
    int wm = wave & 3;        // 0..3  -> 32-row slab
    int wn = wave >> 2;       // 0..1  -> 64-col slab
    int ibase = blockIdx.x * 128;
    int jbase = blockIdx.y * 128;

    unsigned ldsA = (unsigned)(uintptr_t)&As[0];  // low 32 bits = LDS byte offset
    unsigned ldsB = (unsigned)(uintptr_t)&Bs[0];

    v8f acc[2][4];
#pragma unroll
    for (int mt = 0; mt < 2; ++mt)
#pragma unroll
        for (int nt = 0; nt < 4; ++nt)
            acc[mt][nt] = (v8f)0.0f;

    int l15  = lane & 15;
    int lhi  = lane >> 4;          // 0 or 1
    int akoff = lhi * 8;           // A frag: k = akoff..+7 and akoff+16..+23
    int bkoff = lhi * 8;           // B frag: k = bkoff..+7 and bkoff+16..+23

    const int NCH = D / 32;
    bool issuer = (wave == 0);

    // prologue: DMA chunk 0 into buffer 0
    if (issuer) {
        tdm_load_tile_f16(ldsA, xn + (size_t)ibase * D, D, B, 32, 128, D);
        tdm_load_tile_f16(ldsB, xn + (size_t)jbase * D, D, B, 32, 128, D);
    }

    for (int c = 0; c < NCH; ++c) {
        int cur = c & 1;
        if (issuer) __builtin_amdgcn_s_wait_tensorcnt(0);  // buffer `cur` ready
        __syncthreads();  // broadcast readiness; buffer cur^1 now free (compute c-1 done)
        if (issuer && (c + 1 < NCH)) {
            int nxt = cur ^ 1;
            int kc = (c + 1) * 32;
            tdm_load_tile_f16(ldsA + (unsigned)(nxt * TILE_H * 2),
                              xn + (size_t)ibase * D + kc, D, B, 32, 128, D);
            tdm_load_tile_f16(ldsB + (unsigned)(nxt * TILE_H * 2),
                              xn + (size_t)jbase * D + kc, D, B, 32, 128, D);
        }

        const _Float16* Ab = &As[cur * TILE_H];
        const _Float16* Bb = &Bs[cur * TILE_H];

        // load all fragments first, then issue WMMAs as a block
        v16h afrag[2], bfrag[4];
#pragma unroll
        for (int mt = 0; mt < 2; ++mt) {
            int r = wm * 32 + mt * 16 + l15;
            v8h lo = *(const v8h*)&Ab[r * LDS_STRIDE + akoff];
            v8h hi = *(const v8h*)&Ab[r * LDS_STRIDE + akoff + 16];
#pragma unroll
            for (int e = 0; e < 8; ++e) { afrag[mt][e] = lo[e]; afrag[mt][e + 8] = hi[e]; }
        }
#pragma unroll
        for (int nt = 0; nt < 4; ++nt) {
            int n = wn * 64 + nt * 16 + l15;
            v8h lo = *(const v8h*)&Bb[n * LDS_STRIDE + bkoff];
            v8h hi = *(const v8h*)&Bb[n * LDS_STRIDE + bkoff + 16];
#pragma unroll
            for (int e = 0; e < 8; ++e) { bfrag[nt][e] = lo[e]; bfrag[nt][e + 8] = hi[e]; }
        }
#pragma unroll
        for (int nt = 0; nt < 4; ++nt)
#pragma unroll
            for (int mt = 0; mt < 2; ++mt)
                acc[mt][nt] = __builtin_amdgcn_wmma_f32_16x16x32_f16(
                    false, afrag[mt], false, bfrag[nt],
                    (short)0, acc[mt][nt], false, false);
    }

    // epilogue: exp(2*S) and row-sum. C layout: lanes0-15 hold M=r, lanes16-31 M=r+8.
#pragma unroll
    for (int mt = 0; mt < 2; ++mt) {
        float rowpart[8];
#pragma unroll
        for (int r = 0; r < 8; ++r) {
            float s = 0.0f;
#pragma unroll
            for (int nt = 0; nt < 4; ++nt) s += __expf(TEMP_INV * acc[mt][nt][r]);
            rowpart[r] = s;
        }
#pragma unroll
        for (int r = 0; r < 8; ++r) {
#pragma unroll
            for (int off = 1; off < 16; off <<= 1)
                rowpart[r] += __shfl_xor(rowpart[r], off, 32);
        }
        int rowbase = ibase + wm * 32 + mt * 16 + (lhi ? 8 : 0);
        if (l15 == 0) {
#pragma unroll
            for (int r = 0; r < 8; ++r)
                atomicAdd(&total[rowbase + r], rowpart[r]);
        }
    }
}

// ------------------------------------------------------------------ finalize
__global__ void k_final(const float* __restrict__ total, const float* __restrict__ posS,
                        const float* __restrict__ ce_acc, float* __restrict__ out, int B) {
    __shared__ float red[256];
    int tid = threadIdx.x;
    float s = 0.0f;
    for (int i = tid; i < B; i += 256)
        s += __logf(total[i]) - TEMP_INV * posS[i];
    red[tid] = s;
    __syncthreads();
    for (int st = 128; st > 0; st >>= 1) {
        if (tid < st) red[tid] += red[tid + st];
        __syncthreads();
    }
    if (tid == 0) {
        float lf = red[0] / (float)B;
        float la = ce_acc[0] / (float)B;
        out[0] = la + LAMBD * lf;
        out[1] = la;
        out[2] = lf;
    }
}

extern "C" void kernel_launch(void* const* d_in, const int* in_sizes, int n_in,
                              void* d_out, int out_size, void* d_ws, size_t ws_size,
                              hipStream_t stream) {
    const float* data = (const float*)d_in[0];   // [B, C]
    const float* x    = (const float*)d_in[1];   // [B, D]
    const int*   y    = (const int*)d_in[2];     // [B]
    float* out = (float*)d_out;                  // 3 floats

    _Float16* xn    = (_Float16*)d_ws;                         // B*D f16 = 8 MB
    float*    total = (float*)((char*)d_ws + (size_t)B_ROWS * D_DIM * sizeof(_Float16));
    float*    posS  = total + B_ROWS;
    float*    ceacc = posS + B_ROWS;

    int nz = 2 * B_ROWS + 1;
    k_zero<<<(nz + 255) / 256, 256, 0, stream>>>(total, nz);

    k_normalize<<<B_ROWS, 256, 0, stream>>>(x, xn, D_DIM);

    k_pos<<<B_ROWS / 8, 256, 0, stream>>>(xn, y, posS, B_ROWS, D_DIM);

    k_ce<<<B_ROWS, 256, 0, stream>>>(data, y, ceacc, C_DIM);

    dim3 grid(B_ROWS / 128, B_ROWS / 128);
    k_gemm_rowsum<<<grid, 256, 0, stream>>>(xn, total, B_ROWS, D_DIM);

    k_final<<<1, 256, 0, stream>>>(total, posS, ceacc, out, B_ROWS);
}
